// MurelCell_5334349382028
// MI455X (gfx1250) — compile-verified, matrix operand order checked by reading, not run on hardware
//
#include <hip/hip_runtime.h>

typedef __attribute__((ext_vector_type(4)))  float    f32x4;
typedef __attribute__((ext_vector_type(8)))  float    f32x8;
typedef __attribute__((ext_vector_type(4)))  __bf16   bf16x4;
typedef __attribute__((ext_vector_type(8)))  __bf16   bf16x8;
typedef __attribute__((ext_vector_type(16))) __bf16   bf16x16;
typedef __attribute__((ext_vector_type(4)))  unsigned int u32x4;
typedef __attribute__((ext_vector_type(8)))  int      i32x8;
typedef __attribute__((ext_vector_type(4)))  int      i32x4;

#define B_SZ 32
#define N_OBJ 36
#define BNR (B_SZ * N_OBJ)   // 1152 rows
#define QD 4800
#define OD 2048
#define FD 2048
#define AD 512

#if defined(__gfx1250__) && __has_builtin(__builtin_amdgcn_tensor_load_to_lds)
#define HAVE_TDM 1
#else
#define HAVE_TDM 0
#endif

#if HAVE_TDM
// ---------------------------------------------------------------------------
// Tensor Data Mover: 2-D tile load global->LDS (bf16, data_size=2B).
// D# packing per CDNA5 ISA §8.3/§8.4 (normal mode, no pad/iterate/gather).
//   group0: [1:0]count=1 | lds_addr | global_addr(57b) | type=2
//   group1: data_size=1(2B), tensor_dim0/1, tile_dim0/1, tensor_dim0_stride
// tile rows land contiguously in LDS: LDS[row*tile_d0 + k].
// Issue from ONE wave; completion via s_wait_tensorcnt 0 then barrier.
// This toolchain exposes the 6-arg builtin:
//   (u32x4 g0, i32x8 g1, i32x4 g2, i32x4 g3, i32x8 pad, i32 cpol)
// ---------------------------------------------------------------------------
__device__ __forceinline__ void tdm_load_2d(unsigned int lds_off, const void* gptr,
                                            unsigned int tile_d0, unsigned int tile_d1,
                                            unsigned long long d0_stride) {
    unsigned long long ga = (unsigned long long)gptr;
    u32x4 g0;
    g0[0] = 1u;                                    // count=1, user descriptor
    g0[1] = lds_off;                               // LDS byte address
    g0[2] = (unsigned int)ga;                      // global_addr[31:0]
    g0[3] = (unsigned int)(ga >> 32) | (2u << 30); // global_addr[56:32] | type=2
    i32x8 g1;
    g1[0] = (int)(1u << 16);                       // wg_mask=0, data_size=1 (2B)
    g1[1] = (int)((tile_d0 & 0xFFFFu) << 16);      // tensor_dim0[15:0] @ [63:48]
    g1[2] = (int)((tile_d0 >> 16) | ((tile_d1 & 0xFFFFu) << 16));
    g1[3] = (int)((tile_d1 >> 16) | ((tile_d0 & 0xFFFFu) << 16)); // tile_dim0 @ [127:112]
    g1[4] = (int)(tile_d1 & 0xFFFFu);              // tile_dim1 @ [143:128], tile_dim2=0
    g1[5] = (int)(unsigned int)(d0_stride & 0xFFFFFFFFull);
    g1[6] = (int)(unsigned int)((d0_stride >> 32) & 0xFFFFull);
    g1[7] = 0;
    i32x4 z4 = {0, 0, 0, 0};
    i32x8 z8 = {0, 0, 0, 0, 0, 0, 0, 0};
    __builtin_amdgcn_tensor_load_to_lds(g0, g1, z4, z4, z8, 0);
}
__device__ __forceinline__ unsigned int lds_off_of(const void* p) {
    return (unsigned int)(unsigned long long)p;    // LDS aperture: addr[31:0]
}
#endif

// ---------------------------------------------------------------------------
// Fragment loader: 16-bit A/B operand for v_wmma_*_16x16x32 (wave32).
// ISA 7.12.2: lane L(0..15) row L holds K = kk+0..7 then kk+16..23;
// lane L+16 holds K = kk+8..15 then kk+24..31.
// ---------------------------------------------------------------------------
__device__ __forceinline__ bf16x16 frag_ld(const __bf16* base, int stride, int kk) {
    int lane = threadIdx.x & 31;
    const __bf16* p = base + (lane & 15) * stride + kk + ((lane >> 4) << 3);
    bf16x8 lo = *(const bf16x8*)(p);
    bf16x8 hi = *(const bf16x8*)(p + 16);
    return __builtin_shufflevector(lo, hi, 0,1,2,3,4,5,6,7,8,9,10,11,12,13,14,15);
}

// ---------------------------------------------------------------------------
__global__ __launch_bounds__(256) void cvt_kernel(const float* __restrict__ in,
                                                  __bf16* __restrict__ out, int n) {
    int i = blockIdx.x * 256 + threadIdx.x;
    if (i < n) out[i] = (__bf16)in[i];
}

// f32 [K][N] -> bf16 [N][K] transpose+convert
__global__ __launch_bounds__(256) void cvtT_kernel(const float* __restrict__ in,
                                                   __bf16* __restrict__ out,
                                                   int K, int Nn) {
    __shared__ float tile[32][33];
    int k0 = blockIdx.x * 32, n0 = blockIdx.y * 32;
    int tx = threadIdx.x, ty = threadIdx.y;
    for (int r = ty; r < 32; r += 8)
        if (k0 + r < K && n0 + tx < Nn)
            tile[r][tx] = in[(size_t)(k0 + r) * Nn + n0 + tx];
    __syncthreads();
    for (int r = ty; r < 32; r += 8)
        if (n0 + r < Nn && k0 + tx < K)
            out[(size_t)(n0 + r) * K + k0 + tx] = (__bf16)tile[tx][r];
}

// pb1/pb2: boxes[1152,4] @ Wb[4,2048] + bb  (K=4, pure VALU)
__global__ __launch_bounds__(256) void boxproj_kernel(
    const float* __restrict__ boxes,
    const float* __restrict__ Wb1, const float* __restrict__ bb1,
    const float* __restrict__ Wb2, const float* __restrict__ bb2,
    float* __restrict__ pb1, float* __restrict__ pb2) {
    int idx = blockIdx.x * 256 + threadIdx.x;
    if (idx >= BNR * FD) return;
    int r = idx / FD, d = idx % FD;
    const float* bx = boxes + r * 4;
    float s1 = bb1[d], s2 = bb2[d];
#pragma unroll
    for (int c = 0; c < 4; ++c) {
        s1 += bx[c] * Wb1[c * FD + d];
        s2 += bx[c] * Wb2[c * FD + d];
    }
    pb1[idx] = s1;
    pb2[idx] = s2;
}

// ---------------------------------------------------------------------------
// Tiled bf16 WMMA GEMM: C[M,N] = A[M,K] @ Bt[N,K]^T + bias.
// Block 64x128, 8 waves (32x32 each -> 2x2 frags), BK=64.
// Tiles staged via TDM (tensor_load_to_lds) when available.
// ---------------------------------------------------------------------------
template <bool HAS_MULT, bool HAS_CB>
__global__ __launch_bounds__(256) void gemm_bf16_kernel(
    const __bf16* __restrict__ A, const __bf16* __restrict__ Bt,
    const float* __restrict__ bias, const float* __restrict__ mult,
    float* __restrict__ Cf, __bf16* __restrict__ Cb,
    int M, int Nn, int K) {
    __shared__ __bf16 Alds[64 * 64];    // 8 KB
    __shared__ __bf16 Blds[128 * 64];   // 16 KB
    const int tid  = threadIdx.x;
    const int lane = tid & 31;
    const int wave = tid >> 5;
    const int wm = wave & 1, wn = wave >> 1;
    const int mb = blockIdx.y * 64;
    const int nb = blockIdx.x * 128;

    f32x8 zero = {};
    f32x8 acc[2][2];
#pragma unroll
    for (int i = 0; i < 2; ++i)
#pragma unroll
        for (int j = 0; j < 2; ++j) acc[i][j] = zero;

    for (int kc = 0; kc < K; kc += 64) {
        __syncthreads();
#if HAVE_TDM
        if (wave == 0) {
            tdm_load_2d(lds_off_of(Alds), A + (size_t)mb * K + kc, 64, 64,
                        (unsigned long long)K);
            tdm_load_2d(lds_off_of(Blds), Bt + (size_t)nb * K + kc, 64, 128,
                        (unsigned long long)K);
            __builtin_amdgcn_s_wait_tensorcnt(0);
        }
#else
#pragma unroll
        for (int g = 0; g < 2; ++g) {
            int t = tid + g * 256;
            int row = t >> 3, k0 = (t & 7) << 3;
            *(bf16x8*)&Alds[row * 64 + k0] =
                *(const bf16x8*)&A[(size_t)(mb + row) * K + kc + k0];
        }
#pragma unroll
        for (int g = 0; g < 4; ++g) {
            int t = tid + g * 256;
            int row = t >> 3, k0 = (t & 7) << 3;
            *(bf16x8*)&Blds[row * 64 + k0] =
                *(const bf16x8*)&Bt[(size_t)(nb + row) * K + kc + k0];
        }
#endif
        __syncthreads();
#pragma unroll
        for (int kk = 0; kk < 64; kk += 32) {
            bf16x16 afr[2], bfr[2];
#pragma unroll
            for (int mi = 0; mi < 2; ++mi)
                afr[mi] = frag_ld(&Alds[(wm * 32 + mi * 16) * 64], 64, kk);
#pragma unroll
            for (int ni = 0; ni < 2; ++ni)
                bfr[ni] = frag_ld(&Blds[(wn * 32 + ni * 16) * 64], 64, kk);
#pragma unroll
            for (int mi = 0; mi < 2; ++mi)
#pragma unroll
                for (int ni = 0; ni < 2; ++ni)
                    acc[mi][ni] = __builtin_amdgcn_wmma_f32_16x16x32_bf16(
                        false, afr[mi], false, bfr[ni], (short)0, acc[mi][ni],
                        false, false);
        }
    }
    // Epilogue (C layout: N = lane&15, M = v + 8*(lane>>4))
    const int hi = lane >> 4, nl = lane & 15;
#pragma unroll
    for (int mi = 0; mi < 2; ++mi)
#pragma unroll
        for (int ni = 0; ni < 2; ++ni) {
            int col = nb + wn * 32 + ni * 16 + nl;
            float bv = bias[col];
#pragma unroll
            for (int v = 0; v < 8; ++v) {
                int row = mb + wm * 32 + mi * 16 + v + hi * 8;
                size_t idx = (size_t)row * Nn + col;
                float val = acc[mi][ni][v] + bv;
                if (HAS_MULT) val *= mult[idx];
                Cf[idx] = val;
                if (HAS_CB) Cb[idx] = (__bf16)val;
            }
        }
}

// ---------------------------------------------------------------------------
// Scores kernel: one WG per (b,i). Relation rows
//   r_j[k] = pf1[b,i,k]*pf2[b,j,k] + pb1[b,i,k]*pb2[b,j,k]   (j<36, pad 48)
// built per k-chunk in LDS (bf16); [48,2048]@[2048,512] via WMMA;
// scores[b,i,j] = sum_n tanh(H+ba0)*Wa1 + ba1.  Wa0T chunk staged via TDM.
// ---------------------------------------------------------------------------
__global__ __launch_bounds__(256) void scores_kernel(
    const float* __restrict__ pf1, const float* __restrict__ pf2,
    const float* __restrict__ pb1, const float* __restrict__ pb2,
    const __bf16* __restrict__ Wa0T, const float* __restrict__ ba0,
    const float* __restrict__ Wa1, const float* __restrict__ ba1,
    float* __restrict__ scores) {
    __shared__ __bf16 Abuf[48 * 32];    // 3 KB
    __shared__ __bf16 Bbuf[512 * 32];   // 32 KB
    __shared__ float  sc[48];
    const int tid  = threadIdx.x;
    const int lane = tid & 31;
    const int wave = tid >> 5;          // 8 waves x 64 att-cols
    const int bi = blockIdx.x;
    const int b  = bi / N_OBJ;
    const float* p1 = pf1 + (size_t)bi * FD;
    const float* q1 = pb1 + (size_t)bi * FD;

    if (tid < 48) sc[tid] = (tid < 36) ? ba1[0] : 0.f;

    f32x8 zero = {};
    f32x8 acc[3][4];
#pragma unroll
    for (int mt = 0; mt < 3; ++mt)
#pragma unroll
        for (int nt = 0; nt < 4; ++nt) acc[mt][nt] = zero;

    for (int kc = 0; kc < FD; kc += 32) {
        __syncthreads();
#if HAVE_TDM
        if (wave == 0)      // async DMA of Wa0T chunk while all waves build A
            tdm_load_2d(lds_off_of(Bbuf), Wa0T + kc, 32, 512,
                        (unsigned long long)FD);
#endif
        // Relation rows, vectorized: 48 rows x 8 groups of 4 k
        for (int g = tid; g < 48 * 8; g += 256) {
            int j = g >> 3;
            int k = (g & 7) << 2;
            f32x4 r = {};
            if (j < 36) {
                size_t rj = (size_t)(b * N_OBJ + j) * FD + kc + k;
                f32x4 a1 = *(const f32x4*)&p1[kc + k];
                f32x4 b1 = *(const f32x4*)&pf2[rj];
                f32x4 a2 = *(const f32x4*)&q1[kc + k];
                f32x4 b2 = *(const f32x4*)&pb2[rj];
                r = a1 * b1 + a2 * b2;
            }
            *(bf16x4*)&Abuf[j * 32 + k] = __builtin_convertvector(r, bf16x4);
        }
#if HAVE_TDM
        if (wave == 0) __builtin_amdgcn_s_wait_tensorcnt(0);
#else
#pragma unroll
        for (int g = 0; g < 8; ++g) {
            int t = tid + g * 256;
            int n = t >> 2, k0 = (t & 3) << 3;
            *(bf16x8*)&Bbuf[n * 32 + k0] =
                *(const bf16x8*)&Wa0T[(size_t)n * FD + kc + k0];
        }
#endif
        __syncthreads();
        bf16x16 afr[3];
#pragma unroll
        for (int mt = 0; mt < 3; ++mt)
            afr[mt] = frag_ld(&Abuf[mt * 16 * 32], 32, 0);
#pragma unroll
        for (int nt = 0; nt < 4; ++nt) {
            bf16x16 bfr = frag_ld(&Bbuf[(wave * 64 + nt * 16) * 32], 32, 0);
#pragma unroll
            for (int mt = 0; mt < 3; ++mt)
                acc[mt][nt] = __builtin_amdgcn_wmma_f32_16x16x32_bf16(
                    false, afr[mt], false, bfr, (short)0, acc[mt][nt],
                    false, false);
        }
    }

    // tanh + Wa1 dot, reduce across the 512 att columns
    const int hi = lane >> 4, nl = lane & 15;
    float part[3][8];
#pragma unroll
    for (int mt = 0; mt < 3; ++mt)
#pragma unroll
        for (int v = 0; v < 8; ++v) part[mt][v] = 0.f;
#pragma unroll
    for (int nt = 0; nt < 4; ++nt) {
        int n = wave * 64 + nt * 16 + nl;
        float b0 = ba0[n], w1 = Wa1[n];
#pragma unroll
        for (int mt = 0; mt < 3; ++mt)
#pragma unroll
            for (int v = 0; v < 8; ++v)
                part[mt][v] += tanhf(acc[mt][nt][v] + b0) * w1;
    }
#pragma unroll
    for (int mt = 0; mt < 3; ++mt)
#pragma unroll
        for (int v = 0; v < 8; ++v) {
            float x = part[mt][v];
            x += __shfl_xor(x, 1);
            x += __shfl_xor(x, 2);
            x += __shfl_xor(x, 4);
            x += __shfl_xor(x, 8);
            if (nl == 0) atomicAdd(&sc[mt * 16 + v + hi * 8], x);  // ds_add_f32
        }
    __syncthreads();
    if (tid < 36) scores[(size_t)bi * N_OBJ + tid] = sc[tid];
}

// ---------------------------------------------------------------------------
// Finalize: softmax over j + factorized e_hat + residual adds.
// ---------------------------------------------------------------------------
__global__ __launch_bounds__(256) void finalize_kernel(
    const float* __restrict__ scores,
    const float* __restrict__ pf1, const float* __restrict__ pf2,
    const float* __restrict__ pb1, const float* __restrict__ pb2,
    const float* __restrict__ fused, const float* __restrict__ obj,
    float* __restrict__ out) {
    __shared__ float sL[N_OBJ];
    __shared__ float aL[N_OBJ];
    const int tid = threadIdx.x;
    const int bi  = blockIdx.x;
    const int b   = bi / N_OBJ;
    if (tid < N_OBJ) sL[tid] = scores[(size_t)bi * N_OBJ + tid];
    __syncthreads();
    float mx = -1e30f;
    for (int j = 0; j < N_OBJ; ++j) mx = fmaxf(mx, sL[j]);
    float sm = 0.f;
    for (int j = 0; j < N_OBJ; ++j) sm += __expf(sL[j] - mx);
    if (tid < N_OBJ) aL[tid] = __expf(sL[tid] - mx) / sm;
    __syncthreads();
    size_t base = (size_t)bi * FD;
    for (int d = tid; d < FD; d += 256) {
        float s2 = 0.f, s4 = 0.f;
        for (int j = 0; j < N_OBJ; ++j) {
            size_t rj = (size_t)(b * N_OBJ + j) * FD + d;
            float a = aL[j];
            s2 += a * pf2[rj];
            s4 += a * pb2[rj];
        }
        out[base + d] = obj[base + d] + fused[base + d] +
                        pf1[base + d] * s2 + pb1[base + d] * s4;
    }
}

// ---------------------------------------------------------------------------
extern "C" void kernel_launch(void* const* d_in, const int* in_sizes, int n_in,
                              void* d_out, int out_size, void* d_ws, size_t ws_size,
                              hipStream_t stream) {
    const float* qe  = (const float*)d_in[0];
    const float* obj = (const float*)d_in[1];
    const float* box = (const float*)d_in[2];
    const float* Wq  = (const float*)d_in[3];
    const float* bq  = (const float*)d_in[4];
    const float* Wo  = (const float*)d_in[5];
    const float* bo  = (const float*)d_in[6];
    const float* Wf1 = (const float*)d_in[7];
    const float* bf1 = (const float*)d_in[8];
    const float* Wf2 = (const float*)d_in[9];
    const float* bf2 = (const float*)d_in[10];
    const float* Wb1 = (const float*)d_in[11];
    const float* bb1 = (const float*)d_in[12];
    const float* Wb2 = (const float*)d_in[13];
    const float* bb2 = (const float*)d_in[14];
    const float* Wa0 = (const float*)d_in[15];
    const float* ba0 = (const float*)d_in[16];
    const float* Wa1 = (const float*)d_in[17];
    const float* ba1 = (const float*)d_in[18];
    float* out = (float*)d_out;

    char* ws = (char*)d_ws;
    size_t off = 0;
    auto alloc = [&](size_t bytes) -> char* {
        char* p = ws + off;
        off += (bytes + 255) & ~(size_t)255;
        return p;
    };
    __bf16* qeB    = (__bf16*)alloc((size_t)BNR * QD * 2);
    __bf16* WqT    = (__bf16*)alloc((size_t)FD * QD * 2);
    __bf16* objB   = (__bf16*)alloc((size_t)BNR * OD * 2);
    __bf16* WoT    = (__bf16*)alloc((size_t)FD * OD * 2);
    __bf16* Wf1T   = (__bf16*)alloc((size_t)FD * FD * 2);
    __bf16* Wf2T   = (__bf16*)alloc((size_t)FD * FD * 2);
    __bf16* Wa0T   = (__bf16*)alloc((size_t)AD * FD * 2);
    float*  tmpQ   = (float*)alloc((size_t)BNR * FD * 4);
    float*  fusedF = (float*)alloc((size_t)BNR * FD * 4);
    __bf16* fusedB = (__bf16*)alloc((size_t)BNR * FD * 2);
    float*  pf1    = (float*)alloc((size_t)BNR * FD * 4);
    float*  pf2    = (float*)alloc((size_t)BNR * FD * 4);
    float*  pb1    = (float*)alloc((size_t)BNR * FD * 4);
    float*  pb2    = (float*)alloc((size_t)BNR * FD * 4);
    float*  scoresG = (float*)alloc((size_t)BNR * N_OBJ * 4);

    dim3 tb(32, 8);
    {
        int n = BNR * QD;
        cvt_kernel<<<(n + 255) / 256, 256, 0, stream>>>(qe, qeB, n);
    }
    {
        int n = BNR * OD;
        cvt_kernel<<<(n + 255) / 256, 256, 0, stream>>>(obj, objB, n);
    }
    cvtT_kernel<<<dim3(QD / 32, FD / 32), tb, 0, stream>>>(Wq, WqT, QD, FD);
    cvtT_kernel<<<dim3(OD / 32, FD / 32), tb, 0, stream>>>(Wo, WoT, OD, FD);
    cvtT_kernel<<<dim3(FD / 32, FD / 32), tb, 0, stream>>>(Wf1, Wf1T, FD, FD);
    cvtT_kernel<<<dim3(FD / 32, FD / 32), tb, 0, stream>>>(Wf2, Wf2T, FD, FD);
    cvtT_kernel<<<dim3(FD / 32, AD / 32), tb, 0, stream>>>(Wa0, Wa0T, FD, AD);

    dim3 gg(FD / 128, BNR / 64);  // (16, 18)
    gemm_bf16_kernel<false, false><<<gg, 256, 0, stream>>>(
        qeB, WqT, bq, nullptr, tmpQ, nullptr, BNR, FD, QD);
    gemm_bf16_kernel<true, true><<<gg, 256, 0, stream>>>(
        objB, WoT, bo, tmpQ, fusedF, fusedB, BNR, FD, OD);
    gemm_bf16_kernel<false, false><<<gg, 256, 0, stream>>>(
        fusedB, Wf1T, bf1, nullptr, pf1, nullptr, BNR, FD, FD);
    gemm_bf16_kernel<false, false><<<gg, 256, 0, stream>>>(
        fusedB, Wf2T, bf2, nullptr, pf2, nullptr, BNR, FD, FD);
    boxproj_kernel<<<(BNR * FD + 255) / 256, 256, 0, stream>>>(
        box, Wb1, bb1, Wb2, bb2, pb1, pb2);
    scores_kernel<<<BNR, 256, 0, stream>>>(pf1, pf2, pb1, pb2, Wa0T, ba0,
                                           Wa1, ba1, scoresG);
    finalize_kernel<<<BNR, 256, 0, stream>>>(scoresG, pf1, pf2, pb1, pb2,
                                             fusedF, obj, out);
}
